// LigerFusedLinearGRPOLoss_28003186770049
// MI455X (gfx1250) — compile-verified, matrix operand order checked by reading, not run on hardware
//
#include <hip/hip_runtime.h>
#include <stdint.h>

// ---------------- problem constants (from reference) ----------------
#define BB   4
#define TT   512
#define BT   (BB * TT)      // 2048 tokens
#define HD   2048
#define VV   32000
#define BETA 0.04f
#define EPSL 0.2f
#define EPSH 0.2f
#define LOG2E 1.4426950408889634f
#define LN2   0.6931471805599453f

typedef __bf16 bf16;
typedef bf16  v16bf __attribute__((ext_vector_type(16)));
typedef bf16  v8bf  __attribute__((ext_vector_type(8)));
typedef bf16  v4bf  __attribute__((ext_vector_type(4)));
typedef float v8f   __attribute__((ext_vector_type(8)));
typedef float v4f   __attribute__((ext_vector_type(4)));

// ---------------- GEMM tiling ----------------
#define MT 16                 // tokens per block
#define NT 128                // vocab columns per n-tile (8 waves x 16)
#define KC 128                // k-chunk staged in LDS per step
#define NCHUNK (HD / KC)      // 16
#define NTILES (VV / NT)      // 250
#define A_STRIDE (HD * 2 + 16)   // 4112 B, padded (bank de-conflict)
#define B_STRIDE (KC * 2 + 16)   // 272 B, padded
#define A_BYTES (MT * A_STRIDE)  // 65792
#define B_BYTES (NT * B_STRIDE)  // 34816
#define LDS_A   0
#define LDS_B0  (A_BYTES)
#define LDS_RED (A_BYTES + 2 * B_BYTES)       // 135424
#define LDS_TOTAL (LDS_RED + 8 * 16 * 3 * 4)  // + reduce scratch = 136960

// ---------------- CDNA5 async copy: global -> LDS, 16B per lane ----------------
__device__ __forceinline__ void async_copy16(uint32_t lds_off, unsigned long long gbase,
                                             uint32_t goff) {
  // GVS mode: 64-bit SGPR base + 32-bit VGPR byte offset; VDST holds LDS byte address.
  asm volatile("global_load_async_to_lds_b128 %0, %1, %2"
               :: "v"(lds_off), "v"(goff), "s"(gbase) : "memory");
}
__device__ __forceinline__ void wait_async_0() {
  asm volatile("s_wait_asynccnt 0" ::: "memory");
}

__device__ __forceinline__ v8f zero_v8f() {
  v8f z;
#pragma unroll
  for (int i = 0; i < 8; ++i) z[i] = 0.f;
  return z;
}

// Load one 16x32 bf16 WMMA fragment from LDS (ISA 7.12.2 layout):
// per lane: 8 bf16 at base, 8 bf16 at base+32B (K and K+16 groups).
__device__ __forceinline__ v16bf load_frag(const char* smem, int off) {
  v8bf lo = *(const v8bf*)(smem + off);
  v8bf hi = *(const v8bf*)(smem + off + 32);
  return __builtin_shufflevector(lo, hi, 0, 1, 2, 3, 4, 5, 6, 7,
                                 8, 9, 10, 11, 12, 13, 14, 15);
}

// ---------------- kernel 1: fp32 -> bf16 convert ----------------
__global__ void __launch_bounds__(256) convert_f32_bf16(const float* __restrict__ src,
                                                        bf16* __restrict__ dst, int n4) {
  int i = blockIdx.x * blockDim.x + threadIdx.x;
  int stride = gridDim.x * blockDim.x;
  for (; i < n4; i += stride) {
    v4f f = *(const v4f*)(src + 4 * (size_t)i);
    *(v4bf*)(dst + 4 * (size_t)i) = __builtin_convertvector(f, v4bf);
  }
}

// ---------------- kernel 2: fused bf16 GEMM + online logsumexp/entropy ----------------
// grid.x = 2 * (BT/MT) = 256 blocks; block = 256 threads = 8 waves (wave32).
__global__ void __launch_bounds__(256) grpo_lse_kernel(
    const bf16* __restrict__ xb0, const bf16* __restrict__ wb0, const float* __restrict__ bias0,
    const bf16* __restrict__ xb1, const bf16* __restrict__ wb1, const float* __restrict__ bias1,
    float* __restrict__ lse_out, float* __restrict__ ent_out) {
  extern __shared__ char smem[];
  const int tid = threadIdx.x;
  const int lane = tid & 31;
  const int wave = tid >> 5;          // n-subtile index 0..7
  const int prob = blockIdx.x >> 7;   // 0 = policy, 1 = ref
  const int m0 = (blockIdx.x & 127) * MT;

  const bf16* __restrict__ xb = prob ? xb1 : xb0;
  const bf16* __restrict__ wb = prob ? wb1 : wb0;
  const float* __restrict__ bias = prob ? bias1 : bias0;
  float* __restrict__ lse_o = lse_out + prob * BT;
  float* __restrict__ ent_o = ent_out + prob * BT;

  const unsigned long long xg = (unsigned long long)(uintptr_t)xb;
  const unsigned long long wg = (unsigned long long)(uintptr_t)wb;
  // generic pointer low 32 bits == LDS byte offset for shared allocations
  const uint32_t lds_base = (uint32_t)(uintptr_t)smem;

  // ---- prologue: stage A tile (16 x 2048 bf16) into LDS asynchronously ----
#pragma unroll
  for (int it = 0; it < 16; ++it) {
    int u = it * 256 + tid;           // 0..4095 units of 16B
    int row = u >> 8, un = u & 255;
    async_copy16(lds_base + LDS_A + row * A_STRIDE + un * 16,
                 xg, ((uint32_t)(m0 + row) * HD + un * 8) * 2u);
  }

  auto issue_b = [&](int c) {         // stage one 128x128 bf16 W chunk
    int n0 = (c >> 4) * NT;
    int k0 = (c & 15) * KC;
    int buf = c & 1;
#pragma unroll
    for (int it = 0; it < 8; ++it) {
      int u = it * 256 + tid;         // 0..2047 units of 16B
      int col = u >> 4, un = u & 15;
      async_copy16(lds_base + LDS_B0 + buf * B_BYTES + col * B_STRIDE + un * 16,
                   wg, ((uint32_t)(n0 + col) * HD + (uint32_t)k0 + un * 8) * 2u);
    }
  };

  // per-lane online softmax state: 8 rows per lane-half, base-2 domain
  float rm[8], rs[8], ru[8];
#pragma unroll
  for (int i = 0; i < 8; ++i) { rm[i] = -1e30f; rs[i] = 0.f; ru[i] = 0.f; }

  const int half = (lane >> 4) & 1;
  const int colLane = (wave << 4) | (lane & 15);
  const int aRow = LDS_A + (lane & 15) * A_STRIDE;
  const int bCol = (lane & 15) * B_STRIDE;
  const int kHalfB = half * 16;       // 8 bf16 * 2B

  issue_b(0);
  v8f acc = zero_v8f();
  float bval = 0.f;
  const int total = NTILES * NCHUNK;  // 4000 chunks
  for (int c = 0; c < total; ++c) {
    const int kc = c & 15;
    const int buf = c & 1;

    // Publish chunk c: my async loads done, then block-wide barrier. The same
    // barrier also proves every wave finished reading buf[(c+1)&1] (iter c-1),
    // so issuing chunk c+1 right after it is safe -> one barrier per chunk.
    wait_async_0();
    __syncthreads();
    if (c + 1 < total) issue_b(c + 1);

    if (kc == 0) {
      acc = zero_v8f();
      bval = bias[(c >> 4) * NT + colLane];   // hoisted: hides load latency
    }

    // preload all 8 fragments of this chunk, then run the 4 WMMAs so the
    // scheduler can overlap ds_load latency with matrix math
    const int bBase = LDS_B0 + buf * B_BYTES + bCol;
    v16bf af[4], bf[4];
#pragma unroll
    for (int j = 0; j < 4; ++j) {
      af[j] = load_frag(smem, aRow + (kc * KC + j * 32) * 2 + kHalfB);
      bf[j] = load_frag(smem, bBase + (j * 32) * 2 + kHalfB);
    }
#pragma unroll
    for (int j = 0; j < 4; ++j)
      acc = __builtin_amdgcn_wmma_f32_16x16x32_bf16(false, af[j], false, bf[j],
                                                    (short)0, acc, false, false);

    if (kc == 15) {                   // n-tile done: online update, 16 cols/lane-slot
#pragma unroll
      for (int i = 0; i < 8; ++i) {
        float l = acc[i] + bval;      // TEMPERATURE == 1
        float a = l * LOG2E;
        float mo = rm[i];
        float mn = fmaxf(mo, a);
        float p = __builtin_amdgcn_exp2f(a - mn);
        float q = __builtin_amdgcn_exp2f(mo - mn);
        rs[i] = rs[i] * q + p;
        ru[i] = ru[i] * q + l * p;
        rm[i] = mn;
      }
    }
  }

  // ---- reduce across the 16 lanes of each half (columns of a row) ----
#pragma unroll
  for (int i = 0; i < 8; ++i) {
    float m = rm[i], s = rs[i], u = ru[i];
#pragma unroll
    for (int off = 1; off < 16; off <<= 1) {
      float m2 = __shfl_xor(m, off), s2 = __shfl_xor(s, off), u2 = __shfl_xor(u, off);
      float mn = fmaxf(m, m2);
      float e1 = __builtin_amdgcn_exp2f(m - mn), e2 = __builtin_amdgcn_exp2f(m2 - mn);
      s = s * e1 + s2 * e2;
      u = u * e1 + u2 * e2;
      m = mn;
    }
    rm[i] = m; rs[i] = s; ru[i] = u;
  }
  __syncthreads();
  float* red = (float*)(smem + LDS_RED);
  if ((lane & 15) == 0) {
#pragma unroll
    for (int i = 0; i < 8; ++i) {
      int idx = (wave * 16 + (i + 8 * half)) * 3;
      red[idx] = rm[i]; red[idx + 1] = rs[i]; red[idx + 2] = ru[i];
    }
  }
  __syncthreads();
  if (tid < 16) {                     // merge 8 wave partials per row
    float m = -1e30f, s = 0.f, u = 0.f;
    for (int w = 0; w < 8; ++w) {
      int idx = (w * 16 + tid) * 3;
      float m2 = red[idx], s2 = red[idx + 1], u2 = red[idx + 2];
      float mn = fmaxf(m, m2);
      float e1 = __builtin_amdgcn_exp2f(m - mn), e2 = __builtin_amdgcn_exp2f(m2 - mn);
      s = s * e1 + s2 * e2;
      u = u * e1 + u2 * e2;
      m = mn;
    }
    float lse = (m + __builtin_amdgcn_logf(s)) * LN2;  // natural logsumexp
    lse_o[m0 + tid] = lse;
    ent_o[m0 + tid] = lse - u / s;                     // entropy
  }
}

// ---------------- kernel 3: selected-token logit (fp32 dot) ----------------
__global__ void __launch_bounds__(256) sel_logit_kernel(
    const float* __restrict__ x, const float* __restrict__ w, const float* __restrict__ bias,
    const int* __restrict__ sel_ids, float* __restrict__ out) {
  const int tok = blockIdx.x;
  const int s = sel_ids[tok];
  const float* xr = x + (size_t)tok * HD;
  const float* wr = w + (size_t)s * HD;
  float acc = 0.f;
  for (int k = threadIdx.x; k < HD; k += 256) acc += xr[k] * wr[k];
#pragma unroll
  for (int off = 16; off; off >>= 1) acc += __shfl_xor(acc, off);
  __shared__ float sb[8];
  if ((threadIdx.x & 31) == 0) sb[threadIdx.x >> 5] = acc;
  __syncthreads();
  if (threadIdx.x == 0) {
    float t = 0.f;
#pragma unroll
    for (int i = 0; i < 8; ++i) t += sb[i];
    out[tok] = t + bias[s];
  }
}

// ---------------- kernel 4: finalize (4 scalar outputs) ----------------
__global__ void __launch_bounds__(256) finalize_kernel(
    const float* __restrict__ lse, const float* __restrict__ ent,
    const float* __restrict__ selp, const float* __restrict__ mask,
    const float* __restrict__ adv, float* __restrict__ out) {
  __shared__ float acc[5];
  if (threadIdx.x < 5) acc[threadIdx.x] = 0.f;
  __syncthreads();
  float s0 = 0.f, s1 = 0.f, s2 = 0.f, s3 = 0.f, s4 = 0.f;
  for (int t = threadIdx.x; t < BT; t += 256) {
    float mk = mask[t];
    float a = adv[t / TT];
    float logp = selp[t] - lse[t];
    float rlogp = selp[BT + t] - lse[BT + t];
    float d = rlogp - logp;
    float kl = __expf(d) - d - 1.f;
    float coef1 = __expf(logp - logp);  // exactly 1 in forward pass
    float coef2 = fminf(fmaxf(coef1, 1.f - EPSL), 1.f + EPSH);
    float ptl = -fminf(coef1 * a, coef2 * a) + BETA * kl;
    float clip = (((coef1 < 1.f - EPSL) && (a < 0.f)) ||
                  ((coef1 > 1.f + EPSH) && (a > 0.f))) ? 1.f : 0.f;
    s0 += ptl * mk; s1 += kl * mk; s2 += clip * mk; s3 += ent[t] * mk; s4 += mk;
  }
#pragma unroll
  for (int off = 16; off; off >>= 1) {
    s0 += __shfl_xor(s0, off); s1 += __shfl_xor(s1, off); s2 += __shfl_xor(s2, off);
    s3 += __shfl_xor(s3, off); s4 += __shfl_xor(s4, off);
  }
  if ((threadIdx.x & 31) == 0) {
    atomicAdd(&acc[0], s0); atomicAdd(&acc[1], s1); atomicAdd(&acc[2], s2);
    atomicAdd(&acc[3], s3); atomicAdd(&acc[4], s4);
  }
  __syncthreads();
  if (threadIdx.x == 0) {
    float denom = fmaxf(acc[4], 1.f);
    out[0] = acc[0] / denom;   // loss
    out[1] = acc[1] / denom;   // kl metric
    out[2] = acc[2] / denom;   // clip ratio
    out[3] = acc[3] / denom;   // mean entropy
  }
}

// ---------------- host-side launch ----------------
extern "C" void kernel_launch(void* const* d_in, const int* in_sizes, int n_in,
                              void* d_out, int out_size, void* d_ws, size_t ws_size,
                              hipStream_t stream) {
  const float* x    = (const float*)d_in[0];
  const float* w    = (const float*)d_in[1];
  const float* bias = (const float*)d_in[2];
  const int*   sel  = (const int*)d_in[3];
  const float* mask = (const float*)d_in[4];
  const float* adv  = (const float*)d_in[5];
  const float* rx   = (const float*)d_in[6];
  const float* rw   = (const float*)d_in[7];
  const float* rb   = (const float*)d_in[8];
  float* out = (float*)d_out;

  // workspace layout (bf16 copies keep each weight matrix < 192MB L2)
  char* ws = (char*)d_ws;
  size_t off = 0;
  bf16* xb  = (bf16*)(ws + off); off += (size_t)BT * HD * 2;
  bf16* rxb = (bf16*)(ws + off); off += (size_t)BT * HD * 2;
  bf16* wb  = (bf16*)(ws + off); off += (size_t)VV * HD * 2;
  bf16* rwb = (bf16*)(ws + off); off += (size_t)VV * HD * 2;
  float* lse  = (float*)(ws + off); off += 2 * BT * sizeof(float);
  float* ent  = (float*)(ws + off); off += 2 * BT * sizeof(float);
  float* selp = (float*)(ws + off); off += 2 * BT * sizeof(float);
  (void)ws_size; (void)in_sizes; (void)n_in; (void)out_size;

  convert_f32_bf16<<<1024, 256, 0, stream>>>(x,  xb,  BT * HD / 4);
  convert_f32_bf16<<<1024, 256, 0, stream>>>(rx, rxb, BT * HD / 4);
  convert_f32_bf16<<<4096, 256, 0, stream>>>(w,  wb,  VV * HD / 4);
  convert_f32_bf16<<<4096, 256, 0, stream>>>(rw, rwb, VV * HD / 4);

  grpo_lse_kernel<<<2 * (BT / MT), 256, LDS_TOTAL, stream>>>(
      xb, wb, bias, rxb, rwb, rb, lse, ent);

  sel_logit_kernel<<<BT, 256, 0, stream>>>(x,  w,  bias, sel, selp);
  sel_logit_kernel<<<BT, 256, 0, stream>>>(rx, rw, rb,   sel, selp + BT);

  finalize_kernel<<<1, 256, 0, stream>>>(lse, ent, selp, mask, adv, out);
}
